// BANLayer_3582002725425
// MI455X (gfx1250) — matrix operand order repaired and checked by reading, not compiled, not run on hardware
//
#include <hip/hip_runtime.h>

// ---------------------------------------------------------------------------
// Problem constants (from the reference)
// ---------------------------------------------------------------------------
static constexpr int B_    = 32;
static constexpr int NV_   = 256;
static constexpr int NQ_   = 512;
static constexpr int VD_   = 512;   // = QD
static constexpr int HK_   = 1536;  // HD*K = 512*3
static constexpr int HD_   = 512;
static constexpr int KPOOL = 3;
static constexpr int HOUT_ = 8;

typedef __attribute__((ext_vector_type(16))) __bf16          v16bf;
typedef __attribute__((ext_vector_type(8)))  float           v8f;
typedef __attribute__((ext_vector_type(8)))  unsigned short  us8;

// ---------------------------------------------------------------------------
// bf16 helpers: RNE f32->bf16 and hi/lo split for the 3-term bf16 WMMA scheme
// (A*B ~= Ah*Bh + Ah*Bl + Al*Bh, f32 accumulate; ~2^-16 relative error).
// ---------------------------------------------------------------------------
__device__ __forceinline__ unsigned short f2bf_rne(float f) {
  unsigned int u = __float_as_uint(f);
  unsigned int r = u + 0x7FFFu + ((u >> 16) & 1u);
  return (unsigned short)(r >> 16);
}
__device__ __forceinline__ float bf2f(unsigned short h) {
  return __uint_as_float(((unsigned int)h) << 16);
}
__device__ __forceinline__ void split_bf(float x, unsigned short& hi, unsigned short& lo) {
  hi = f2bf_rne(x);
  lo = f2bf_rne(x - bf2f(hi));
}

// Pack two adjacent (hi,lo) splits and store as single b32 LDS stores.
__device__ __forceinline__ void store_pair(unsigned short* Hi, unsigned short* Lo, int idx,
                                           float x0, float x1) {
  unsigned short h0, l0, h1, l1;
  split_bf(x0, h0, l0);
  split_bf(x1, h1, l1);
  *(unsigned int*)(Hi + idx) = (unsigned int)h0 | ((unsigned int)h1 << 16);
  *(unsigned int*)(Lo + idx) = (unsigned int)l0 | ((unsigned int)l1 << 16);
}

// Build a v16bf fragment from two 16-byte-aligned 8xu16 LDS chunks
// (guarantees ds_load_b128 vectorization).
__device__ __forceinline__ v16bf frag2(const unsigned short* p0, const unsigned short* p1) {
  struct { us8 a, b; } tmp;
  tmp.a = *(const us8*)p0;
  tmp.b = *(const us8*)p1;
  v16bf r;
  __builtin_memcpy(&r, &tmp, 32);
  return r;
}

// ---------------------------------------------------------------------------
// LDS tile geometry: 64x64 output macro-tile, K-step 32, 8 waves x 2 subtiles,
// double-buffered. Operands stored as [row][k]; row stride 40 ushorts (80 B)
// -> conflict-free across 16 lanes, all chunks 16-B aligned.
// ---------------------------------------------------------------------------
static constexpr int BK   = 32;
static constexpr int ASTR = BK + 8;     // 40 ushorts
static constexpr int TSZ  = 64 * ASTR;  // one buffer

// Wave-level 3-term bf16 WMMA step, CDNA5 layouts (ISA 7.12.2):
//  A (16x32 MxK): lane L -> row L%16; elem i -> K=(i/8)*16+(i%8)+8*(L/16)
//  B (32x16 KxN): lane L -> col L%16; elem i -> K=(i/8)*16+(i%8)+8*(L/16)
//  C/D: lane L, vgpr r -> row r+8*(L/16), col L%16
__device__ __forceinline__ v8f wmma3_step(const unsigned short* Ah, const unsigned short* Al,
                                          const unsigned short* Bh, const unsigned short* Bl,
                                          int tm, int tn, int lane, v8f acc) {
  const int l = lane & 15, half = lane >> 4;
  const int arow = (tm * 16 + l) * ASTR;
  const int brow = (tn * 16 + l) * ASTR;
  v16bf ah = frag2(Ah + arow + 8 * half, Ah + arow + 16 + 8 * half);
  v16bf al = frag2(Al + arow + 8 * half, Al + arow + 16 + 8 * half);
  v16bf bh = frag2(Bh + brow + 8 * half, Bh + brow + 16 + 8 * half);
  v16bf bl = frag2(Bl + brow + 8 * half, Bl + brow + 16 + 8 * half);
  acc = __builtin_amdgcn_wmma_f32_16x16x32_bf16(false, al, false, bh, (short)0, acc, false, false);
  acc = __builtin_amdgcn_wmma_f32_16x16x32_bf16(false, ah, false, bl, (short)0, acc, false, false);
  acc = __builtin_amdgcn_wmma_f32_16x16x32_bf16(false, ah, false, bh, (short)0, acc, false, false);
  return acc;
}

// ---------------------------------------------------------------------------
// Stage 0: weight-norm scale = g / ||V||_F (deterministic fixed-order partials)
// ---------------------------------------------------------------------------
__global__ __launch_bounds__(256) void sumsq_part_kernel(const float* __restrict__ x, int n,
                                                         float* __restrict__ part) {
  __shared__ float red[256];
  float s = 0.f;
  for (int i = blockIdx.x * 256 + threadIdx.x; i < n; i += 256 * 256) {
    float t = x[i];
    s += t * t;
  }
  red[threadIdx.x] = s;
  __syncthreads();
  for (int w = 128; w > 0; w >>= 1) {
    if (threadIdx.x < w) red[threadIdx.x] += red[threadIdx.x + w];
    __syncthreads();
  }
  if (threadIdx.x == 0) part[blockIdx.x] = red[0];
}

__global__ __launch_bounds__(256) void norm_finalize_kernel(const float* __restrict__ vg,
                                                            const float* __restrict__ qg,
                                                            float* __restrict__ sc) {
  __shared__ float r0[256], r1[256];
  const int t = threadIdx.x;
  r0[t] = sc[16 + t];
  r1[t] = sc[16 + 256 + t];
  __syncthreads();
  for (int w = 128; w > 0; w >>= 1) {
    if (t < w) { r0[t] += r0[t + w]; r1[t] += r1[t + w]; }
    __syncthreads();
  }
  if (t == 0) {
    sc[2] = vg[0] / sqrtf(r0[0]);
    sc[3] = qg[0] / sqrtf(r1[0]);
  }
}

// ---------------------------------------------------------------------------
// Stage 1: projections  out = relu(scale * (X @ W^T) + bias)
//   X: (M x 512), W: (1536 x 512) row-major, out: (M x 1536).
//   WRITE_T (v path, 256 rows per batch): also writes outT[b][n][vloc].
//   Double-buffered pipeline: globals for step+1 issued before WMMAs of step.
//   grid = (1536/64, M/64), block = 256.
// ---------------------------------------------------------------------------
template <bool WRITE_T>
__global__ __launch_bounds__(256) void proj_kernel(const float* __restrict__ X,
                                                   const float* __restrict__ W,
                                                   const float* __restrict__ bias,
                                                   const float* __restrict__ scale_p,
                                                   float* __restrict__ outF,
                                                   float* __restrict__ outT) {
  constexpr int N = HK_, K = VD_, NSTEP = K / BK;
  __shared__ alignas(16) unsigned short Ah[2 * TSZ], Al[2 * TSZ];
  __shared__ alignas(16) unsigned short Bh[2 * TSZ], Bl[2 * TSZ];
  const int tid = threadIdx.x, wave = tid >> 5, lane = tid & 31;
  const int row0 = blockIdx.y * 64, col0 = blockIdx.x * 64;

  float2 ra[4], rb[4];
  auto loadG = [&](int k0) {
#pragma unroll
    for (int i = 0; i < 4; ++i) {
      const int p = tid + i * 256;
      const int r = p >> 4, kk = (p & 15) * 2;
      ra[i] = *(const float2*)&X[(size_t)(row0 + r) * K + k0 + kk];
      rb[i] = *(const float2*)&W[(size_t)(col0 + r) * K + k0 + kk];  // B[n][k] = W[n][k]
    }
  };
  auto storeL = [&](int buf) {
#pragma unroll
    for (int i = 0; i < 4; ++i) {
      const int p = tid + i * 256;
      const int r = p >> 4, kk = (p & 15) * 2;
      store_pair(Ah + buf * TSZ, Al + buf * TSZ, r * ASTR + kk, ra[i].x, ra[i].y);
      store_pair(Bh + buf * TSZ, Bl + buf * TSZ, r * ASTR + kk, rb[i].x, rb[i].y);
    }
  };

  v8f acc[2] = {};
  loadG(0);
  storeL(0);
  for (int step = 0; step < NSTEP; ++step) {
    __syncthreads();
    if (step + 1 < NSTEP) loadG((step + 1) * BK);
    const int buf = step & 1;
#pragma unroll
    for (int wt = 0; wt < 2; ++wt) {
      const int st = wave * 2 + wt, tm = st >> 2, tn = st & 3;
      acc[wt] = wmma3_step(Ah + buf * TSZ, Al + buf * TSZ, Bh + buf * TSZ, Bl + buf * TSZ,
                           tm, tn, lane, acc[wt]);
    }
    if (step + 1 < NSTEP) storeL((step + 1) & 1);
  }

  const float s = *scale_p;
  const int l = lane & 15, half = lane >> 4;
#pragma unroll
  for (int wt = 0; wt < 2; ++wt) {
    const int st = wave * 2 + wt, tm = st >> 2, tn = st & 3;
    const int n = col0 + tn * 16 + l;
    const float bv = bias[n];
#pragma unroll
    for (int r = 0; r < 8; ++r) {
      const int m = row0 + tm * 16 + r + 8 * half;
      const float val = fmaxf(fmaf(s, acc[wt][r], bv), 0.f);
      outF[(size_t)m * N + n] = val;
      if (WRITE_T)  // v path: b = m>>8, vloc = m&255
        outT[((size_t)(m >> 8) * HK_ + n) * NV_ + (m & 255)] = val;
    }
  }
}

// ---------------------------------------------------------------------------
// Stage 2: att[b,h,v,q] = sum_k hm[h,k]*v_[b,v,k]*q_[b,q,k] + hb[h]
//   A = hm[h] ⊙ v_[b] (256x1536), B rows = q_ rows (contiguous k).
//   grid = (512/64, 256/64, B*HOUT), block = 256.
// ---------------------------------------------------------------------------
__global__ __launch_bounds__(256) void att_kernel(const float* __restrict__ vf,
                                                  const float* __restrict__ qf,
                                                  const float* __restrict__ hm,
                                                  const float* __restrict__ hb,
                                                  float* __restrict__ att) {
  constexpr int NSTEP = HK_ / BK;
  __shared__ alignas(16) unsigned short Ah[2 * TSZ], Al[2 * TSZ];
  __shared__ alignas(16) unsigned short Bh[2 * TSZ], Bl[2 * TSZ];
  const int tid = threadIdx.x, wave = tid >> 5, lane = tid & 31;
  const int z = blockIdx.z, b = z >> 3, h = z & 7;
  const int v0 = blockIdx.y * 64, q0 = blockIdx.x * 64;
  const float* Vb  = vf + (size_t)b * NV_ * HK_;
  const float* Qb  = qf + (size_t)b * NQ_ * HK_;
  const float* hmr = hm + (size_t)h * HK_;

  float2 ra[4], rb[4];
  auto loadG = [&](int k0) {
#pragma unroll
    for (int i = 0; i < 4; ++i) {
      const int p = tid + i * 256;
      const int r = p >> 4, kk = (p & 15) * 2;
      const float2 hw = *(const float2*)&hmr[k0 + kk];
      const float2 xa = *(const float2*)&Vb[(size_t)(v0 + r) * HK_ + k0 + kk];
      ra[i] = make_float2(xa.x * hw.x, xa.y * hw.y);
      rb[i] = *(const float2*)&Qb[(size_t)(q0 + r) * HK_ + k0 + kk];
    }
  };
  auto storeL = [&](int buf) {
#pragma unroll
    for (int i = 0; i < 4; ++i) {
      const int p = tid + i * 256;
      const int r = p >> 4, kk = (p & 15) * 2;
      store_pair(Ah + buf * TSZ, Al + buf * TSZ, r * ASTR + kk, ra[i].x, ra[i].y);
      store_pair(Bh + buf * TSZ, Bl + buf * TSZ, r * ASTR + kk, rb[i].x, rb[i].y);
    }
  };

  v8f acc[2] = {};
  loadG(0);
  storeL(0);
  for (int step = 0; step < NSTEP; ++step) {
    __syncthreads();
    if (step + 1 < NSTEP) loadG((step + 1) * BK);
    const int buf = step & 1;
#pragma unroll
    for (int wt = 0; wt < 2; ++wt) {
      const int st = wave * 2 + wt, tm = st >> 2, tn = st & 3;
      acc[wt] = wmma3_step(Ah + buf * TSZ, Al + buf * TSZ, Bh + buf * TSZ, Bl + buf * TSZ,
                           tm, tn, lane, acc[wt]);
    }
    if (step + 1 < NSTEP) storeL((step + 1) & 1);
  }

  const float bias = hb[h];
  const int l = lane & 15, half = lane >> 4;
#pragma unroll
  for (int wt = 0; wt < 2; ++wt) {
    const int st = wave * 2 + wt, tm = st >> 2, tn = st & 3;
    const int qn = q0 + tn * 16 + l;
#pragma unroll
    for (int r = 0; r < 8; ++r) {
      const int vr = v0 + tm * 16 + r + 8 * half;
      att[((size_t)z * NV_ + vr) * NQ_ + qn] = acc[wt][r] + bias;
    }
  }
}

// ---------------------------------------------------------------------------
// Stage 3: asumT[b,q,v] = sum_h att[b,h,v,q]   (LDS-tiled transpose; coalesced
// reads along q, writes along v).  grid = (NQ/64, NV/64, B), block = 256.
// ---------------------------------------------------------------------------
__global__ __launch_bounds__(256) void attsumT_kernel(const float* __restrict__ att,
                                                      float* __restrict__ asumT) {
  __shared__ float T[64][65];
  const int b = blockIdx.z, v0 = blockIdx.y * 64, q0 = blockIdx.x * 64;
#pragma unroll
  for (int i = 0; i < 16; ++i) {
    const int e = threadIdx.x + i * 256;
    const int vl = e >> 6, ql = e & 63;
    float s = 0.f;
#pragma unroll
    for (int h = 0; h < HOUT_; ++h)
      s += att[(((size_t)b * HOUT_ + h) * NV_ + v0 + vl) * NQ_ + q0 + ql];
    T[vl][ql] = s;
  }
  __syncthreads();
#pragma unroll
  for (int i = 0; i < 16; ++i) {
    const int e = threadIdx.x + i * 256;
    const int ql = e >> 6, vl = e & 63;
    asumT[((size_t)b * NQ_ + q0 + ql) * NV_ + v0 + vl] = T[vl][ql];
  }
}

// ---------------------------------------------------------------------------
// Stage 4: u[b,q,k] = sum_v asumT[b,q,v]*vT[b,k,v]  (contraction over v, both
// operands contiguous in v). Fused epilogue: p = sum_r u[q,k]*q_[b,q,k] into
// 64 fixed slots per (b,k) -> deterministic, no atomics.
//   grid = (1536/64, 512/64, B), block = 256.
// ---------------------------------------------------------------------------
__global__ __launch_bounds__(256) void fusion_kernel(const float* __restrict__ asumT,
                                                     const float* __restrict__ vT,
                                                     const float* __restrict__ qf,
                                                     float* __restrict__ fpart) {
  constexpr int NSTEP = NV_ / BK;
  __shared__ alignas(16) unsigned short Ah[2 * TSZ], Al[2 * TSZ];
  __shared__ alignas(16) unsigned short Bh[2 * TSZ], Bl[2 * TSZ];
  const int tid = threadIdx.x, wave = tid >> 5, lane = tid & 31;
  const int b = blockIdx.z, q0 = blockIdx.y * 64, n0 = blockIdx.x * 64;
  const float* Ab = asumT + (size_t)b * NQ_ * NV_;
  const float* Bb = vT + (size_t)b * HK_ * NV_;

  float2 ra[4], rb[4];
  auto loadG = [&](int k0) {
#pragma unroll
    for (int i = 0; i < 4; ++i) {
      const int p = tid + i * 256;
      const int r = p >> 4, kk = (p & 15) * 2;
      ra[i] = *(const float2*)&Ab[(size_t)(q0 + r) * NV_ + k0 + kk];
      rb[i] = *(const float2*)&Bb[(size_t)(n0 + r) * NV_ + k0 + kk];
    }
  };
  auto storeL = [&](int buf) {
#pragma unroll
    for (int i = 0; i < 4; ++i) {
      const int p = tid + i * 256;
      const int r = p >> 4, kk = (p & 15) * 2;
      store_pair(Ah + buf * TSZ, Al + buf * TSZ, r * ASTR + kk, ra[i].x, ra[i].y);
      store_pair(Bh + buf * TSZ, Bl + buf * TSZ, r * ASTR + kk, rb[i].x, rb[i].y);
    }
  };

  v8f acc[2] = {};
  loadG(0);
  storeL(0);
  for (int step = 0; step < NSTEP; ++step) {
    __syncthreads();
    if (step + 1 < NSTEP) loadG((step + 1) * BK);
    const int buf = step & 1;
#pragma unroll
    for (int wt = 0; wt < 2; ++wt) {
      const int st = wave * 2 + wt, tm = st >> 2, tn = st & 3;
      acc[wt] = wmma3_step(Ah + buf * TSZ, Al + buf * TSZ, Bh + buf * TSZ, Bl + buf * TSZ,
                           tm, tn, lane, acc[wt]);
    }
    if (step + 1 < NSTEP) storeL((step + 1) & 1);
  }

  const int l = lane & 15, half = lane >> 4;
#pragma unroll
  for (int wt = 0; wt < 2; ++wt) {
    const int st = wave * 2 + wt, tm = st >> 2, tn = st & 3;
    const int n = n0 + tn * 16 + l;
    float p = 0.f;
#pragma unroll
    for (int r = 0; r < 8; ++r) {
      const int qr = q0 + tm * 16 + r + 8 * half;
      p += acc[wt][r] * qf[((size_t)b * NQ_ + qr) * HK_ + n];
    }
    const int slot = blockIdx.y * 8 + tm * 2 + half;  // 0..63, unique per (b,n) writer
    fpart[((size_t)b * 64 + slot) * HK_ + n] = p;
  }
}

__global__ __launch_bounds__(256) void fusion_reduce_kernel(const float* __restrict__ fpart,
                                                            float* __restrict__ fus) {
  const int idx = blockIdx.x * 256 + threadIdx.x;  // over B*HK
  const int b = idx / HK_, n = idx - b * HK_;
  const float* p = fpart + (size_t)b * 64 * HK_ + n;
  float s = 0.f;
#pragma unroll
  for (int j = 0; j < 64; ++j) s += p[(size_t)j * HK_];
  fus[idx] = s;
}

// ---------------------------------------------------------------------------
// Stage 5: logits[b,d] = sum_{j<3} fusion[b,3d+j]; train-mode BatchNorm over b.
// ---------------------------------------------------------------------------
__global__ __launch_bounds__(512) void bn_kernel(const float* __restrict__ fus,
                                                 const float* __restrict__ gamma,
                                                 const float* __restrict__ beta,
                                                 float* __restrict__ logits) {
  const int d = threadIdx.x;
  float lg[B_];
  float mu = 0.f;
#pragma unroll
  for (int b = 0; b < B_; ++b) {
    const float* f = fus + (size_t)b * HK_ + d * KPOOL;
    lg[b] = f[0] + f[1] + f[2];
    mu += lg[b];
  }
  mu *= (1.f / B_);
  float var = 0.f;
#pragma unroll
  for (int b = 0; b < B_; ++b) {
    const float t = lg[b] - mu;
    var += t * t;
  }
  var *= (1.f / B_);
  const float inv = rsqrtf(var + 1e-5f) * gamma[d];
  const float bt = beta[d];
#pragma unroll
  for (int b = 0; b < B_; ++b) logits[b * HD_ + d] = (lg[b] - mu) * inv + bt;
}

// ---------------------------------------------------------------------------
// Host-side orchestration
// ---------------------------------------------------------------------------
extern "C" void kernel_launch(void* const* d_in, const int* in_sizes, int n_in,
                              void* d_out, int out_size, void* d_ws, size_t ws_size,
                              hipStream_t stream) {
  const float* v   = (const float*)d_in[0];
  const float* q   = (const float*)d_in[1];
  const float* v_V = (const float*)d_in[2];
  const float* v_g = (const float*)d_in[3];
  const float* v_b = (const float*)d_in[4];
  const float* q_V = (const float*)d_in[5];
  const float* q_g = (const float*)d_in[6];
  const float* q_b = (const float*)d_in[7];
  const float* hm  = (const float*)d_in[8];
  const float* hb  = (const float*)d_in[9];
  const float* gam = (const float*)d_in[10];
  const float* bet = (const float*)d_in[11];

  // Workspace layout (floats)
  float* ws = (float*)d_ws;
  const size_t SC_SZ  = 1024;                             // scales + reduction partials
  const size_t V_OFF  = SC_SZ;                            // v_   : 32*256*1536
  const size_t VT_OFF = V_OFF  + (size_t)B_ * NV_ * HK_;  // v_^T : 32*1536*256
  const size_t Q_OFF  = VT_OFF + (size_t)B_ * HK_ * NV_;  // q_   : 32*512*1536
  const size_t AS_OFF = Q_OFF  + (size_t)B_ * NQ_ * HK_;  // asumT: 32*512*256
  const size_t FP_OFF = AS_OFF + (size_t)B_ * NQ_ * NV_;  // fpart: 32*64*1536
  const size_t FU_OFF = FP_OFF + (size_t)B_ * 64 * HK_;   // fusion: 32*1536
  float* vfeat  = ws + V_OFF;
  float* vfeatT = ws + VT_OFF;
  float* qfeat  = ws + Q_OFF;
  float* asumT  = ws + AS_OFF;
  float* fpart  = ws + FP_OFF;
  float* fus    = ws + FU_OFF;

  float* logits = (float*)d_out;             // 32*512
  float* att    = (float*)d_out + B_ * HD_;  // 32*8*256*512

  // Weight-norm scales (deterministic two-pass reduction)
  sumsq_part_kernel<<<256, 256, 0, stream>>>(v_V, HK_ * VD_, ws + 16);
  sumsq_part_kernel<<<256, 256, 0, stream>>>(q_V, HK_ * VD_, ws + 16 + 256);
  norm_finalize_kernel<<<1, 256, 0, stream>>>(v_g, q_g, ws);

  // Projections (WMMA bf16-split GEMMs); v path also emits transposed copy
  proj_kernel<true><<<dim3(HK_ / 64, (B_ * NV_) / 64), 256, 0, stream>>>(v, v_V, v_b, ws + 2,
                                                                         vfeat, vfeatT);
  proj_kernel<false><<<dim3(HK_ / 64, (B_ * NQ_) / 64), 256, 0, stream>>>(q, q_V, q_b, ws + 3,
                                                                          qfeat, nullptr);

  // Bilinear attention maps -> d_out
  att_kernel<<<dim3(NQ_ / 64, NV_ / 64, B_ * HOUT_), 256, 0, stream>>>(vfeat, qfeat, hm, hb, att);

  // fusion = einsum('bvk,bhvq,bqk->bk'), contracted over v then q
  attsumT_kernel<<<dim3(NQ_ / 64, NV_ / 64, B_), 256, 0, stream>>>(att, asumT);
  fusion_kernel<<<dim3(HK_ / 64, NQ_ / 64, B_), 256, 0, stream>>>(asumT, vfeatT, qfeat, fpart);
  fusion_reduce_kernel<<<(B_ * HK_) / 256, 256, 0, stream>>>(fpart, fus);

  // Pool + BatchNorm -> logits
  bn_kernel<<<1, HD_, 0, stream>>>(fus, gam, bet, logits);
}
